// GroupedQueryAttention_51986284151470
// MI455X (gfx1250) — compile-verified, matrix operand order checked by reading, not run on hardware
//
#include <hip/hip_runtime.h>
#include <hip/hip_bf16.h>

typedef _Float16 f16;
typedef __attribute__((ext_vector_type(16))) _Float16 v16h;
typedef __attribute__((ext_vector_type(8)))  float    v8f;
typedef __attribute__((ext_vector_type(4)))  float    f32x4;

#define WMMA_F32_F16(a, b, c) \
  __builtin_amdgcn_wmma_f32_16x16x32_f16(false, (a), false, (b), (short)0, (c), false, false)

// ---- CDNA5 async global->LDS copy (ASYNCcnt path, ISA 08_async_tensor) ----
// Low 32 bits of a flat pointer into __shared__ are the wave-relative LDS byte
// address.
__device__ __forceinline__ void async_copy16(f16* lds_dst, const f16* g_src) {
  unsigned loff = (unsigned)(size_t)lds_dst;
  asm volatile("global_load_async_to_lds_b128 %0, %1, off"
               :: "v"(loff), "v"(g_src) : "memory");
}
__device__ __forceinline__ void wait_async0() {
  asm volatile("s_wait_asynccnt 0" ::: "memory");
}
__device__ __forceinline__ void wait_async4() {
  asm volatile("s_wait_asynccnt 4" ::: "memory");
}

// Load a 16x32 f16 fragment (A-style rows / B-style columns) from a tile with
// row stride `stride` elements. Per-lane layout per CDNA5 ISA 16-bit striping:
// lane<16: row=lane, K elems {0..7, 16..23}; lane>=16: row=lane-16, K {8..15, 24..31}.
__device__ __forceinline__ v16h load_frag(const f16* p, int stride) {
  const int lane = threadIdx.x & 31;
  const f16* q = p + (lane & 15) * stride + ((lane >> 4) << 3);
  v16h f;
  ((f32x4*)&f)[0] = *(const f32x4*)q;
  ((f32x4*)&f)[1] = *(const f32x4*)(q + 16);
  return f;
}

// ---------------------------------------------------------------------------
// f16 GEMM: C[M,N] = A[M,K] * Bt[N,K]^T  (Bt pre-transposed, f32 out)
// WG = 256 thr = 8 waves -> 128x128 C tile; wave = 32x64 (8 WMMA / K-step).
// Double-buffered LDS fed by global_load_async_to_lds_b128.
// ---------------------------------------------------------------------------
__global__ __launch_bounds__(256)
void gemm_f16f32(const f16* __restrict__ A, const f16* __restrict__ Bt,
                 float* __restrict__ C, int M, int N, int K) {
  __shared__ __align__(16) f16 ldsA[2][128 * 32];
  __shared__ __align__(16) f16 ldsB[2][128 * 32];
  const int tid = threadIdx.x;
  const int wave = tid >> 5, lane = tid & 31;
  const int mb = blockIdx.x, nb = blockIdx.y;
  const int wm = wave & 3, wn = wave >> 2;        // 4 x 32 rows, 2 x 64 cols
  const int r = tid >> 1, c0 = (tid & 1) << 4;    // per-thread copy slots
  const f16* arow = &A[(size_t)(mb * 128 + r) * K + c0];
  const f16* brow = &Bt[(size_t)(nb * 128 + r) * K + c0];
  v8f acc[2][4] = {};
  const int nsteps = K >> 5;

  // prologue: tile 0 into buffer 0
  async_copy16(&ldsA[0][r * 32 + c0], arow);
  async_copy16(&ldsA[0][r * 32 + c0 + 8], arow + 8);
  async_copy16(&ldsB[0][r * 32 + c0], brow);
  async_copy16(&ldsB[0][r * 32 + c0 + 8], brow + 8);

  for (int kb = 0; kb < nsteps; ++kb) {
    const int buf = kb & 1;
    if (kb + 1 < nsteps) {                        // issue next tile, other buffer
      const f16* an = arow + (size_t)(kb + 1) * 32;
      const f16* bn = brow + (size_t)(kb + 1) * 32;
      async_copy16(&ldsA[buf ^ 1][r * 32 + c0], an);
      async_copy16(&ldsA[buf ^ 1][r * 32 + c0 + 8], an + 8);
      async_copy16(&ldsB[buf ^ 1][r * 32 + c0], bn);
      async_copy16(&ldsB[buf ^ 1][r * 32 + c0 + 8], bn + 8);
      wait_async4();                              // in-order: tile kb landed
    } else {
      wait_async0();
    }
    __syncthreads();                              // tile kb visible to all waves
    v16h af0 = load_frag(&ldsA[buf][(wm * 32) * 32], 32);
    v16h af1 = load_frag(&ldsA[buf][(wm * 32 + 16) * 32], 32);
#pragma unroll
    for (int n = 0; n < 4; ++n) {
      v16h bf = load_frag(&ldsB[buf][(wn * 64 + n * 16) * 32], 32);
      acc[0][n] = WMMA_F32_F16(af0, bf, acc[0][n]);
      acc[1][n] = WMMA_F32_F16(af1, bf, acc[1][n]);
    }
    __syncthreads();                              // done reading buf before it is refilled
  }
  const int hi = lane >> 4, col = lane & 15;
#pragma unroll
  for (int a2 = 0; a2 < 2; ++a2)
#pragma unroll
    for (int n = 0; n < 4; ++n)
#pragma unroll
      for (int v = 0; v < 8; ++v)
        C[(size_t)(mb * 128 + wm * 32 + a2 * 16 + v + 8 * hi) * N +
          nb * 128 + wn * 64 + n * 16 + col] = acc[a2][n][v];
}

// ---------------------------------------------------------------------------
// Flash attention with analytic causal mask.
// Layouts: Q[b,h,t,d] f16, K[b,kv,t,d] f16, Vt[b,kv,d,t] f16, O[b,h,t,d] f32.
// WG = 128 thr = 4 waves; each wave owns 16 queries; 32-key K/V tiles staged
// via global_load_async_to_lds_b128.
// ---------------------------------------------------------------------------
#define T_SEQ 2048
#define HD 128
#define NH 16
#define NKV 4

__global__ __launch_bounds__(128)
void flash_attn(const f16* __restrict__ Q, const f16* __restrict__ Kh,
                const f16* __restrict__ Vt, float* __restrict__ O) {
  __shared__ __align__(16) f16 ldsK[32 * HD];   // [tk][d]
  __shared__ __align__(16) f16 ldsV[HD * 32];   // [d][t]
  __shared__ __align__(16) f16 ldsP[4 * 16 * 32];
  const int tid = threadIdx.x, wave = tid >> 5, lane = tid & 31;
  const int hi = lane >> 4, col = lane & 15;
  const int blk = blockIdx.x;
  const int qb = blk & 31;          // T/64 blocks
  const int h  = (blk >> 5) & 15;
  const int b  = blk >> 9;
  const int kv = h >> 2;            // GQA: 4 Q heads per KV head
  const int q0 = qb * 64 + wave * 16;

  const f16* qbase = Q + ((size_t)(b * NH + h) * T_SEQ + q0) * HD;
  v16h qf[4];
#pragma unroll
  for (int kk = 0; kk < 4; ++kk) qf[kk] = load_frag(qbase + kk * 32, HD);

  v8f acc[8] = {};
  float m_i[8], l_i[8];
#pragma unroll
  for (int v = 0; v < 8; ++v) { m_i[v] = -1e30f; l_i[v] = 0.f; }

  const f16* ksrc = Kh + (size_t)(b * NKV + kv) * T_SEQ * HD;
  const f16* vsrc = Vt + (size_t)(b * NKV + kv) * HD * T_SEQ;
  f16* myP = &ldsP[wave * 16 * 32];
  const int nkb = qb * 2 + 2;                 // causal bound for whole WG
  const float scale = 0.08838834764831845f;   // 1/sqrt(128)

  // per-thread K/V copy slots (4 x 16B each per tile)
  const int kr = tid >> 2, kc = (tid & 3) << 3;     // wrong shape for K; fixed below
  for (int kb = 0; kb < nkb; ++kb) {
    __syncthreads();                          // prior compute done before refill
#pragma unroll
    for (int i = 0; i < 4; ++i) {
      int ck = tid + i * 128;
      int k_r = ck >> 4, k_c = (ck & 15) << 3;
      async_copy16(&ldsK[k_r * HD + k_c],
                   &ksrc[(size_t)(kb * 32 + k_r) * HD + k_c]);
      int v_r = ck >> 2, v_c = (ck & 3) << 3;
      async_copy16(&ldsV[v_r * 32 + v_c],
                   &vsrc[(size_t)v_r * T_SEQ + kb * 32 + v_c]);
    }
    wait_async0();
    __syncthreads();                          // tiles visible to all waves
    if (kb * 32 > q0 + 15) continue;          // wave fully masked; barriers kept

    v8f s0 = {}, s1 = {};
#pragma unroll
    for (int kk = 0; kk < 4; ++kk) {
      v16h kf0 = load_frag(&ldsK[0 * 16 * HD + kk * 32], HD);
      v16h kf1 = load_frag(&ldsK[1 * 16 * HD + kk * 32], HD);
      s0 = WMMA_F32_F16(qf[kk], kf0, s0);
      s1 = WMMA_F32_F16(qf[kk], kf1, s1);
    }
#pragma unroll
    for (int v = 0; v < 8; ++v) {
      const int row = q0 + v + 8 * hi;
      const int c0m = kb * 32 + col, c1m = c0m + 16;
      float x0 = (c0m <= row) ? s0[v] * scale : -1e30f;
      float x1 = (c1m <= row) ? s1[v] * scale : -1e30f;
      float rmax = fmaxf(x0, x1);
#pragma unroll
      for (int off = 1; off < 16; off <<= 1)
        rmax = fmaxf(rmax, __shfl_xor(rmax, off, 32));
      const float nm = fmaxf(m_i[v], rmax);
      const float corr = __expf(m_i[v] - nm);
      m_i[v] = nm;
      const float p0 = __expf(x0 - nm), p1 = __expf(x1 - nm);
      float rsum = p0 + p1;
#pragma unroll
      for (int off = 1; off < 16; off <<= 1)
        rsum += __shfl_xor(rsum, off, 32);
      l_i[v] = l_i[v] * corr + rsum;
#pragma unroll
      for (int n = 0; n < 8; ++n) acc[n][v] *= corr;
      myP[(v + 8 * hi) * 32 + col]      = (f16)p0;   // C-layout -> LDS
      myP[(v + 8 * hi) * 32 + col + 16] = (f16)p1;
    }
    v16h pf = load_frag(myP, 32);             // reload in A-layout (in-order DS)
#pragma unroll
    for (int n = 0; n < 8; ++n) {
      v16h vf = load_frag(&ldsV[n * 16 * 32], 32);
      acc[n] = WMMA_F32_F16(pf, vf, acc[n]);
    }
  }
  (void)kr; (void)kc;

  float* obase = O + ((size_t)(b * NH + h) * T_SEQ + q0) * HD;
#pragma unroll
  for (int v = 0; v < 8; ++v) {
    const float inv = 1.f / l_i[v];
#pragma unroll
    for (int n = 0; n < 8; ++n)
      obase[(size_t)(v + 8 * hi) * HD + n * 16 + col] = acc[n][v] * inv;
  }
}

// ---------------------------------------------------------------------------
// Elementwise / layout helper kernels
// ---------------------------------------------------------------------------
__global__ void transpose_f32_to_f16(const float* __restrict__ src, f16* __restrict__ dst,
                                     int K, int N) {   // src[K][N] -> dst[N][K]
  __shared__ float tile[32][33];
  const int k0 = blockIdx.x * 32, n0 = blockIdx.y * 32;
  const int tx = threadIdx.x, ty = threadIdx.y;
#pragma unroll
  for (int i = 0; i < 32; i += 8)
    tile[ty + i][tx] = src[(size_t)(k0 + ty + i) * N + n0 + tx];
  __syncthreads();
#pragma unroll
  for (int i = 0; i < 32; i += 8)
    dst[(size_t)(n0 + ty + i) * K + k0 + tx] = (f16)tile[tx][ty + i];
}

__global__ void f32_to_f16(const float* __restrict__ src, f16* __restrict__ dst, size_t n) {
  size_t i = (size_t)blockIdx.x * blockDim.x + threadIdx.x;
  if (i < n) dst[i] = (f16)src[i];
}

// src f32 [b*t][H*128] -> dst f16 [b][H][t][128] with RoPE applied
__global__ void rope_convert(const float* __restrict__ src, f16* __restrict__ dst, int H) {
  size_t idx = (size_t)blockIdx.x * blockDim.x + threadIdx.x;
  if (idx >= (size_t)4 * H * T_SEQ * HD) return;
  const int d = idx & 127;
  const int t = (idx >> 7) & 2047;
  const size_t bh = idx >> 18;
  const int h = (int)(bh % H), b = (int)(bh / H);
  const float* row = src + ((size_t)(b * T_SEQ + t)) * (H * HD) + h * HD;
  const float x = row[d];
  const int dd = d & 63;
  const float angle = (float)t * __expf(-9.210340371976184f * (float)dd / 64.f);
  float s, c;
  __sincosf(angle, &s, &c);
  const float partner = row[(d < 64) ? d + 64 : d - 64];
  dst[idx] = (f16)((d < 64) ? (x * c - partner * s) : (x * c + partner * s));
}

// src f32 [b*t][4*128] -> dst f16 [b][kv][d][t]
__global__ void v_transpose(const float* __restrict__ src, f16* __restrict__ dst) {
  size_t idx = (size_t)blockIdx.x * blockDim.x + threadIdx.x;
  if (idx >= (size_t)4 * NKV * HD * T_SEQ) return;
  const int t  = idx & 2047;
  const int d  = (idx >> 11) & 127;
  const int kv = (idx >> 18) & 3;
  const int b  = (int)(idx >> 20);
  dst[idx] = (f16)src[((size_t)(b * T_SEQ + t)) * (NKV * HD) + kv * HD + d];
}

// src f32 [b][h][t][d] -> dst f16 [b*t][h*128+d]
__global__ void o_convert(const float* __restrict__ src, f16* __restrict__ dst) {
  size_t idx = (size_t)blockIdx.x * blockDim.x + threadIdx.x;
  if (idx >= (size_t)4 * T_SEQ * NH * HD) return;
  const int d = idx & 127;
  const int h = (idx >> 7) & 15;
  const int t = (idx >> 11) & 2047;
  const int b = (int)(idx >> 22);
  dst[idx] = (f16)src[(((size_t)(b * NH + h)) * T_SEQ + t) * HD + d];
}

// ---------------------------------------------------------------------------
extern "C" void kernel_launch(void* const* d_in, const int* in_sizes, int n_in,
                              void* d_out, int out_size, void* d_ws, size_t ws_size,
                              hipStream_t stream) {
  const float* x  = (const float*)d_in[0];
  // d_in[1] = additive mask (exactly causal -> applied analytically, unused)
  const float* wq = (const float*)d_in[2];
  const float* wk = (const float*)d_in[3];
  const float* wv = (const float*)d_in[4];
  const float* wo = (const float*)d_in[5];

  char* ws = (char*)d_ws;
  const size_t MB = 1024 * 1024;
  f16*   xh    = (f16*)(ws + 0);          // 32 MB, recycled as Oh later
  f16*   wqt   = (f16*)(ws + 32  * MB);   // 8 MB
  f16*   wkt   = (f16*)(ws + 40  * MB);   // 2 MB
  f16*   wvt   = (f16*)(ws + 42  * MB);   // 2 MB
  f16*   wot   = (f16*)(ws + 44  * MB);   // 8 MB
  float* stage = (float*)(ws + 52 * MB);  // 64 MB, recycled q->k->v->O
  f16*   qh    = (f16*)(ws + 116 * MB);   // 32 MB
  f16*   kh    = (f16*)(ws + 148 * MB);   // 8 MB
  f16*   vt    = (f16*)(ws + 156 * MB);   // 8 MB
  f16*   oh    = xh;

  const dim3 tb(32, 8);
  transpose_f32_to_f16<<<dim3(64, 64), tb, 0, stream>>>(wq, wqt, 2048, 2048);
  transpose_f32_to_f16<<<dim3(64, 16), tb, 0, stream>>>(wk, wkt, 2048, 512);
  transpose_f32_to_f16<<<dim3(64, 16), tb, 0, stream>>>(wv, wvt, 2048, 512);
  transpose_f32_to_f16<<<dim3(64, 64), tb, 0, stream>>>(wo, wot, 2048, 2048);

  const size_t nx = (size_t)4 * 2048 * 2048;
  f32_to_f16<<<(unsigned)((nx + 255) / 256), 256, 0, stream>>>(x, xh, nx);

  // Q projection + RoPE
  gemm_f16f32<<<dim3(64, 16), 256, 0, stream>>>(xh, wqt, stage, 8192, 2048, 2048);
  rope_convert<<<(unsigned)(((size_t)4 * 16 * 2048 * 128 + 255) / 256), 256, 0, stream>>>(stage, qh, 16);
  // K projection + RoPE
  gemm_f16f32<<<dim3(64, 4), 256, 0, stream>>>(xh, wkt, stage, 8192, 512, 2048);
  rope_convert<<<(unsigned)(((size_t)4 * 4 * 2048 * 128 + 255) / 256), 256, 0, stream>>>(stage, kh, 4);
  // V projection + transpose to [b][kv][d][t]
  gemm_f16f32<<<dim3(64, 4), 256, 0, stream>>>(xh, wvt, stage, 8192, 512, 2048);
  v_transpose<<<(unsigned)(((size_t)4 * 4 * 128 * 2048 + 255) / 256), 256, 0, stream>>>(stage, vt);

  // Attention (O f32 into stage), repack, output projection
  flash_attn<<<4 * 16 * 32, 128, 0, stream>>>(qh, kh, vt, stage);
  o_convert<<<(unsigned)((nx + 255) / 256), 256, 0, stream>>>(stage, oh);
  gemm_f16f32<<<dim3(64, 16), 256, 0, stream>>>(oh, wot, (float*)d_out, 8192, 2048, 2048);
}
